// VectorQuantizer_29549374996659
// MI455X (gfx1250) — matrix operand order repaired and checked by reading, not compiled, not run on hardware
//
#include <hip/hip_runtime.h>
#include <hip/hip_bf16.h>
#include <math.h>

#define N_PTS   65536
#define K_CODES 1024
#define D_DIM   256
#define BETA_C  0.25f
#define EPS_C   1e-10f

#define ROWS_PER_BLOCK 128           // 8 waves x 16 rows
#define THREADS_MAIN   256

// Set to 0 to fall back to plain global->VGPR->LDS staging.
#define USE_ASYNC_LDS 1

typedef __attribute__((ext_vector_type(16))) _Float16 v16h;
typedef __attribute__((ext_vector_type(4)))  _Float16 v4h;
typedef __attribute__((ext_vector_type(8)))  float    v8f;

// ---------------------------------------------------------------------------
// Prep: codebook fp32 -> f16 workspace, e2[k] = ||e_k||^2, zero counts.
// grid = K_CODES blocks x 64 threads (each thread: 4 consecutive elements)
// ---------------------------------------------------------------------------
__global__ void __launch_bounds__(64) vq_prep(const float* __restrict__ emb,
                                              _Float16* __restrict__ emb16,
                                              float* __restrict__ e2,
                                              float* __restrict__ counts) {
  __shared__ float red[64];
  const int k = blockIdx.x;
  const int t = threadIdx.x;
  const float4 v = reinterpret_cast<const float4*>(emb + (size_t)k * D_DIM)[t];
  v4h h;
  h[0] = (_Float16)v.x; h[1] = (_Float16)v.y;
  h[2] = (_Float16)v.z; h[3] = (_Float16)v.w;
  reinterpret_cast<v4h*>(emb16 + (size_t)k * D_DIM)[t] = h;
  red[t] = v.x * v.x + v.y * v.y + v.z * v.z + v.w * v.w;
  __syncthreads();
  for (int o = 32; o > 0; o >>= 1) {
    if (t < o) red[t] += red[t + o];
    __syncthreads();
  }
  if (t == 0) {
    e2[k] = red[0];
    counts[k] = 0.0f;
  }
}

// ---------------------------------------------------------------------------
// Main: fused WMMA score GEMM + argmin + gather + loss partial.
// grid = N/128 blocks x 256 threads (8 waves x 16 rows each)
// ---------------------------------------------------------------------------
__global__ void __launch_bounds__(THREADS_MAIN) vq_main(
    const float* __restrict__ z,
    const float* __restrict__ emb,
    const _Float16* __restrict__ emb16,
    const float* __restrict__ e2,
    float* __restrict__ counts,
    float* __restrict__ partials,
    float* __restrict__ out) {
  // 16 codes x 256 halves, padded to 264 halves/row (528B) -> conflict-free b128 ds reads
  __shared__ __align__(16) _Float16 sB[16 * 264];
  __shared__ int   sIdx[ROWS_PER_BLOCK];
  __shared__ float sRed[THREADS_MAIN];

  const int tid  = threadIdx.x;
  const int w    = tid >> 5;        // wave id 0..7
  const int lane = tid & 31;
  const int m    = lane & 15;       // row-in-tile / col-in-tile
  const int half = lane >> 4;       // lane half selects K sub-block

  const int rowBase = blockIdx.x * ROWS_PER_BLOCK + w * 16;
  const float* zr = z + (size_t)(rowBase + m) * D_DIM;

  // --- preload A fragments (this wave's 16 rows, all of D=256) ---
  // 16-bit A 16x32 layout: lane<16 holds K {0..7, 16..23}, lane>=16 holds {8..15, 24..31}
  v16h aF[8];
#pragma unroll
  for (int kd = 0; kd < 8; ++kd) {
    const int d0 = kd * 32;
    const float4 x0 = *reinterpret_cast<const float4*>(zr + d0 + half * 8);
    const float4 x1 = *reinterpret_cast<const float4*>(zr + d0 + half * 8 + 4);
    const float4 x2 = *reinterpret_cast<const float4*>(zr + d0 + 16 + half * 8);
    const float4 x3 = *reinterpret_cast<const float4*>(zr + d0 + 16 + half * 8 + 4);
    v16h a;
    a[0]  = (_Float16)x0.x; a[1]  = (_Float16)x0.y; a[2]  = (_Float16)x0.z; a[3]  = (_Float16)x0.w;
    a[4]  = (_Float16)x1.x; a[5]  = (_Float16)x1.y; a[6]  = (_Float16)x1.z; a[7]  = (_Float16)x1.w;
    a[8]  = (_Float16)x2.x; a[9]  = (_Float16)x2.y; a[10] = (_Float16)x2.z; a[11] = (_Float16)x2.w;
    a[12] = (_Float16)x3.x; a[13] = (_Float16)x3.y; a[14] = (_Float16)x3.z; a[15] = (_Float16)x3.w;
    aF[kd] = a;
  }

  float best[8];
  int   bestIdx[8];
#pragma unroll
  for (int r = 0; r < 8; ++r) { best[r] = 3.4e38f; bestIdx[r] = 0; }

  // --- sweep all K=1024 codes, 16 per tile ---
  for (int c = 0; c < K_CODES / 16; ++c) {
    __syncthreads();
    // cooperative stage: 16 codes x 256 halves = 512 x 16B chunks
#pragma unroll
    for (int i = 0; i < 2; ++i) {
      const int chunk = tid + i * THREADS_MAIN;
      const int rr  = chunk >> 5;
      const int col = (chunk & 31) * 8;
      const _Float16* gsrc = emb16 + (size_t)(c * 16 + rr) * D_DIM + col;
      _Float16* ldst = &sB[rr * 264 + col];
#if USE_ASYNC_LDS
      // GLOBAL_LOAD_ASYNC_TO_LDS_B128: VDST = per-lane LDS byte offset,
      // VADDR = 64-bit global address. Tracked by ASYNCcnt.
      {
        const unsigned long long ga = (unsigned long long)gsrc;
        const unsigned lo = (unsigned)(unsigned long long)ldst;  // low 32b = LDS offset
        asm volatile("global_load_async_to_lds_b128 %0, %1, off"
                     :
                     : "v"(lo), "v"(ga)
                     : "memory");
      }
#else
      *reinterpret_cast<uint4*>(ldst) = *reinterpret_cast<const uint4*>(gsrc);
#endif
    }
#if USE_ASYNC_LDS
    asm volatile("s_wait_asynccnt 0x0" ::: "memory");
#endif
    __syncthreads();

    // --- 8 WMMAs with double-buffered B fragments (ds loads overlap matrix pipe) ---
    const _Float16* bbase = &sB[m * 264];   // B: col = lane%16, K = {half*8..+7, 16+half*8..+7}
    v16h bc;
#pragma unroll
    for (int j = 0; j < 16; ++j) bc[j] = bbase[(j < 8 ? j : j + 8) + half * 8];
    v8f acc = {};
#pragma unroll
    for (int kd = 0; kd < 8; ++kd) {
      v16h bn = bc;
      if (kd < 7) {
        const _Float16* bp = bbase + (kd + 1) * 32;
#pragma unroll
        for (int j = 0; j < 16; ++j) bn[j] = bp[(j < 8 ? j : j + 8) + half * 8];
      }
      acc = __builtin_amdgcn_wmma_f32_16x16x32_f16(
          /*neg_a=*/false, aF[kd], /*neg_b=*/false, bc,
          /*c_mod=*/(short)0, acc, /*reuse_a=*/false, /*reuse_b=*/false);
      bc = bn;
    }

    const int   code = c * 16 + m;
    const float e2v  = e2[code];
#pragma unroll
    for (int r = 0; r < 8; ++r) {
      const float dist = fmaf(-2.0f, acc[r], e2v);   // ||e||^2 - 2 z.e
      if (dist < best[r]) { best[r] = dist; bestIdx[r] = code; }
    }
  }

  // --- per-row argmin across the 16 lanes of each half (C layout: col = lane%16) ---
#pragma unroll
  for (int r = 0; r < 8; ++r) {
    float v  = best[r];
    int   bi = bestIdx[r];
#pragma unroll
    for (int off = 8; off >= 1; off >>= 1) {
      const float ov = __shfl_xor(v, off, 16);
      const int   oi = __shfl_xor(bi, off, 16);
      if (ov < v || (ov == v && oi < bi)) { v = ov; bi = oi; }
    }
    if (m == 0) {
      const int localRow = w * 16 + half * 8 + r;   // C rows: r + 8*half
      sIdx[localRow] = bi;
      const int grow = blockIdx.x * ROWS_PER_BLOCK + localRow;
      out[(size_t)N_PTS * D_DIM + 2 + grow] = (float)bi;  // min_encoding_indices
      atomicAdd(&counts[bi], 1.0f);
    }
  }
  __syncthreads();

  // --- gather fp32 codebook rows: z_q_st = z + (z_q - z); accumulate loss ---
  float lsum = 0.0f;
  for (int i = tid; i < ROWS_PER_BLOCK * 64; i += THREADS_MAIN) {  // rows x 64 float4
    const int rr   = i >> 6;
    const int col  = (i & 63) * 4;
    const int code = sIdx[rr];
    const int grow = blockIdx.x * ROWS_PER_BLOCK + rr;
    const float4 e4 = *reinterpret_cast<const float4*>(emb + (size_t)code * D_DIM + col);
    const float4 z4 = *reinterpret_cast<const float4*>(z + (size_t)grow * D_DIM + col);
    const float dx = e4.x - z4.x, dy = e4.y - z4.y, dz = e4.z - z4.z, dw = e4.w - z4.w;
    lsum += dx * dx + dy * dy + dz * dz + dw * dw;
    float4 o4;
    o4.x = z4.x + dx; o4.y = z4.y + dy; o4.z = z4.z + dz; o4.w = z4.w + dw;
    *reinterpret_cast<float4*>(out + (size_t)grow * D_DIM + col) = o4;
  }
  sRed[tid] = lsum;
  __syncthreads();
  for (int o = THREADS_MAIN / 2; o > 0; o >>= 1) {
    if (tid < o) sRed[tid] += sRed[tid + o];
    __syncthreads();
  }
  if (tid == 0) partials[blockIdx.x] = sRed[0];
}

// ---------------------------------------------------------------------------
// Final: loss + perplexity. 1 block x 1024 threads.
// ---------------------------------------------------------------------------
__global__ void __launch_bounds__(1024) vq_final(const float* __restrict__ counts,
                                                 const float* __restrict__ partials,
                                                 float* __restrict__ out) {
  __shared__ float sL[1024];
  __shared__ float sP[1024];
  const int t = threadIdx.x;
  const float cme = counts[t] * (1.0f / (float)N_PTS);
  sP[t] = cme * logf(cme + EPS_C);
  sL[t] = (t < (N_PTS / ROWS_PER_BLOCK)) ? partials[t] : 0.0f;
  __syncthreads();
  for (int o = 512; o > 0; o >>= 1) {
    if (t < o) { sL[t] += sL[t + o]; sP[t] += sP[t + o]; }
    __syncthreads();
  }
  if (t == 0) {
    out[(size_t)N_PTS * D_DIM]     = (1.0f + BETA_C) * sL[0] / ((float)N_PTS * (float)D_DIM);
    out[(size_t)N_PTS * D_DIM + 1] = expf(-sP[0]);
  }
}

// ---------------------------------------------------------------------------
extern "C" void kernel_launch(void* const* d_in, const int* in_sizes, int n_in,
                              void* d_out, int out_size, void* d_ws, size_t ws_size,
                              hipStream_t stream) {
  const float* z   = (const float*)d_in[0];   // (N, D) fp32
  const float* emb = (const float*)d_in[1];   // (K, D) fp32
  float* out = (float*)d_out;                 // [N*D zq | loss | perp | N indices]

  char* ws = (char*)d_ws;
  _Float16* emb16  = (_Float16*)ws;                              // K*D*2   = 512 KB
  float*    e2     = (float*)(ws + (size_t)K_CODES * D_DIM * 2); // K*4     =   4 KB
  float*    counts = e2 + K_CODES;                               // K*4     =   4 KB
  float*    parts  = counts + K_CODES;                           // 512*4   =   2 KB

  vq_prep<<<K_CODES, 64, 0, stream>>>(emb, emb16, e2, counts);
  vq_main<<<N_PTS / ROWS_PER_BLOCK, THREADS_MAIN, 0, stream>>>(z, emb, emb16, e2,
                                                               counts, parts, out);
  vq_final<<<1, 1024, 0, stream>>>(counts, parts, out);
}
